// KernelFeatureMapper_65481071403553
// MI455X (gfx1250) — compile-verified, matrix operand order checked by reading, not run on hardware
//
#include <hip/hip_runtime.h>

// RBF feature map: out[b][r] = exp(-(||x_b||^2 + ||ref_r||^2 - 2 x_b . ref_r))
// x: [16384, 64] f32, refs: [2048, 64] f32, out: [16384, 2048] f32.
//
// MI455X reasoning: output write stream (134 MB) dominates -> ~5.8us at 23.3 TB/s.
// Use V_WMMA_F32_16X16X32_F16 with a hi/lo f16 split of the f32 inputs
// (cross = hi*hi + hi*lo + lo*hi) for ~f32 accuracy on the fast matrix path.
// Inputs (4.7 MB) live in the 192 MB L2; B-fragments are held in registers for
// the wave's lifetime, so per-output-tile L2 traffic is just the A tile.

typedef __attribute__((ext_vector_type(16))) _Float16 v16h;
typedef __attribute__((ext_vector_type(8)))  float    v8f;

__device__ __forceinline__ v8f wmma16x16x32(v16h a, v16h b, v8f c) {
  // (neg_a, A, neg_b, B, c_mod, C, reuse_a, reuse_b)
  return __builtin_amdgcn_wmma_f32_16x16x32_f16(false, a, false, b, (short)0, c,
                                                false, false);
}

// Convert 8 consecutive f32 into hi/lo f16 fragment elements [eo, eo+8),
// accumulating sum-of-squares on the fly.
__device__ __forceinline__ void cvt8(const float* __restrict__ p, v16h& hi,
                                     v16h& lo, int eo, float& ssq) {
#pragma unroll
  for (int j = 0; j < 8; ++j) {
    float v = p[j];
    ssq = fmaf(v, v, ssq);
    _Float16 h = (_Float16)v;
    hi[eo + j] = h;
    lo[eo + j] = (_Float16)(v - (float)h);
  }
}

// A/B fragment layout for 16-bit 16x16x32 WMMA (wave32):
// lane L: row/col = L%16, half = L/16; VGPRs 0..3 hold K = half*8 + {0..7},
// VGPRs 4..7 hold K = 16 + half*8 + {0..7}  (ext-vector elements 0..7, 8..15).
__device__ __forceinline__ void load_half(const float* __restrict__ row, int kh,
                                          int half, v16h& hi, v16h& lo,
                                          float& ssq) {
  const int k0 = kh * 32 + half * 8;
  cvt8(row + k0,      hi, lo, 0, ssq);
  cvt8(row + k0 + 16, hi, lo, 8, ssq);
}

__global__ __launch_bounds__(256) void
rbf_wmma_kernel(const float* __restrict__ x, const float* __restrict__ refs,
                float* __restrict__ out, int colgroups, int R) {
  const int lane = threadIdx.x & 31;
  const int wave = threadIdx.x >> 5;
  const int w = blockIdx.x * 8 + wave;       // global wave id (uniform per wave)
  const int colgroup = w % colgroups;        // owns 32 output columns
  const int rowchunk = w / colgroups;        // owns 128 output rows
  const int nl   = lane & 15;
  const int half = lane >> 4;
  const int nbase = colgroup * 32;

  // ---- Load B fragments once (held in registers for the wave's lifetime) ----
  const float* r0 = refs + (size_t)(nbase + nl) * 64;
  const float* r1 = refs + (size_t)(nbase + 16 + nl) * 64;

  v16h bh[2][2], bl[2][2];                   // [col-tile][k-half]
  float rsq0 = 0.f, rsq1 = 0.f;
  load_half(r0, 0, half, bh[0][0], bl[0][0], rsq0);
  load_half(r0, 1, half, bh[0][1], bl[0][1], rsq0);
  load_half(r1, 0, half, bh[1][0], bl[1][0], rsq1);
  load_half(r1, 1, half, bh[1][1], bl[1][1], rsq1);
  // Each lane summed 32 of the 64 k's; partner half-wave has the rest.
  rsq0 += __shfl_xor(rsq0, 16, 32);          // lane holds ||ref_{nbase+nl}||^2
  rsq1 += __shfl_xor(rsq1, 16, 32);

  // ---- Stream 8 M-tiles (128 rows) against the resident B fragments ----
#pragma unroll 1
  for (int it = 0; it < 8; ++it) {
    const int mbase = rowchunk * 128 + it * 16;
    const float* xr = x + (size_t)(mbase + nl) * 64;

    v16h ah[2], al[2];
    float xsq = 0.f;
    load_half(xr, 0, half, ah[0], al[0], xsq);
    load_half(xr, 1, half, ah[1], al[1], xsq);
    xsq += __shfl_xor(xsq, 16, 32);          // lane L holds ||x_{mbase+L%16}||^2

    v8f c0 = {}, c1 = {};
#pragma unroll
    for (int kh = 0; kh < 2; ++kh) {
      // split-f16 compensated product: hi*hi + hi*lo + lo*hi ~ f32 accuracy
      c0 = wmma16x16x32(ah[kh], bh[0][kh], c0);
      c0 = wmma16x16x32(ah[kh], bl[0][kh], c0);
      c0 = wmma16x16x32(al[kh], bh[0][kh], c0);
      c1 = wmma16x16x32(ah[kh], bh[1][kh], c1);
      c1 = wmma16x16x32(ah[kh], bl[1][kh], c1);
      c1 = wmma16x16x32(al[kh], bh[1][kh], c1);
    }

    // D layout: VGPR i, lanes 0-15 -> M=i, N=lane; lanes 16-31 -> M=i+8, N=lane-16
#pragma unroll
    for (int i = 0; i < 8; ++i) {
      const float xs = __shfl(xsq, i + 8 * half, 32);
      const float d0 = xs + rsq0 - 2.0f * c0[i];
      const float d1 = xs + rsq1 - 2.0f * c1[i];
      const size_t row = (size_t)(mbase + i + 8 * half);
      __builtin_nontemporal_store(__expf(-d0),
                                  out + row * (size_t)R + (size_t)(nbase + nl));
      __builtin_nontemporal_store(__expf(-d1),
                                  out + row * (size_t)R + (size_t)(nbase + 16 + nl));
    }
  }
}

extern "C" void kernel_launch(void* const* d_in, const int* in_sizes, int n_in,
                              void* d_out, int out_size, void* d_ws, size_t ws_size,
                              hipStream_t stream) {
  (void)n_in; (void)d_ws; (void)ws_size; (void)out_size;
  const float* x    = (const float*)d_in[0];
  const float* refs = (const float*)d_in[1];
  float* out = (float*)d_out;

  const int D = 64;
  const int B = in_sizes[0] / D;     // 16384
  const int R = in_sizes[1] / D;     // 2048
  const int colgroups = R / 32;      // 64  (32 output cols per wave)
  const int rowchunks = B / 128;     // 128 (128 output rows per wave)
  const int waves  = colgroups * rowchunks;  // 8192
  const int blocks = waves / 8;              // 1024 blocks x 256 threads (8 waves)

  rbf_wmma_kernel<<<blocks, 256, 0, stream>>>(x, refs, out, colgroups, R);
}
